// GalarModel_87840671138406
// MI455X (gfx1250) — matrix operand order repaired and checked by reading, not compiled, not run on hardware
//
#include <hip/hip_runtime.h>

// ---------------- types ----------------
typedef _Float16 f16;
typedef _Float16 v8h  __attribute__((ext_vector_type(8)));
typedef _Float16 v16h __attribute__((ext_vector_type(16)));
typedef float    v8f  __attribute__((ext_vector_type(8)));

#define SHUF16(a,b) __builtin_shufflevector((a),(b),0,1,2,3,4,5,6,7,8,9,10,11,12,13,14,15)

// ---------------- model dims ----------------
#define FEATD  1024
#define HIDD   512
#define NHEADS 8
#define WINSZ  64
#define TLEN   1024
#define BATCH  4
#define GCND   256
#define DTRANK 32
#define NSTATE 16
#define MROWS  (BATCH*TLEN)   // 4096
#define NANAT  8
#define NPATH  9

// GEMM tile config
#define BM 128        // rows per workgroup (8 waves x 16)
#define BN 64         // cols per workgroup (4 wmma tiles per wave)
#define BKH 64        // K-step (halves) = two wmma k-slices

// ---------------- device helpers ----------------
__device__ __forceinline__ float sigmf(float x){ return 1.0f/(1.0f+__expf(-x)); }
__device__ __forceinline__ float qgeluf(float x){ return x*sigmf(1.702f*x); }
__device__ __forceinline__ float siluf(float x){ return x*sigmf(x); }
__device__ __forceinline__ float softplusf_(float x){ return (x > 20.0f) ? x : log1pf(__expf(x)); }

// blockDim.x must be 256
__device__ __forceinline__ float block_reduce_sum(float v, float* sm){
  sm[threadIdx.x] = v;
  __syncthreads();
  for (int s = 128; s > 0; s >>= 1){
    if ((int)threadIdx.x < s) sm[threadIdx.x] += sm[threadIdx.x + s];
    __syncthreads();
  }
  float r = sm[0];
  __syncthreads();
  return r;
}

__device__ __forceinline__ v8f wmma16(v16h a, v16h b, v8f c){
  return __builtin_amdgcn_wmma_f32_16x16x32_f16(false, a, false, b, (short)0, c, false, false);
}

// low 32 bits of a generic pointer to a __shared__ object == LDS byte offset (ISA 10.2)
__device__ __forceinline__ unsigned lds_off(const void* p){
  return (unsigned)(unsigned long long)p;
}

// async global -> LDS 16B copy (GV mode), tracked by ASYNCcnt
__device__ __forceinline__ void async_ld_b128(unsigned ldsaddr, const void* gaddr){
  asm volatile("global_load_async_to_lds_b128 %0, %1, off"
               :: "v"(ldsaddr), "v"((unsigned long long)gaddr) : "memory");
}
__device__ __forceinline__ void wait_asynccnt0(){
  asm volatile("s_wait_asynccnt 0" ::: "memory");
}

// ================= double-buffered LDS-tiled WMMA GEMM =================
// out[m,n] = act( sum_k A16[m, k] * W16[n, kofs+k] + bias[n] + res[m,n] )
// A16: row-major [Mrows x lda] (per batch, stride aB elements)
// W16: row-major [N x ldb]     (per batch, stride bB elements)
// Workgroup tile: BM x BN; K pipelined in BKH steps through 2 LDS buffers staged
// with GLOBAL_LOAD_ASYNC_TO_LDS_B128 (ASYNCcnt), overlapping copy with WMMA.
// ACT: 0 = none, 1 = qgelu, 3 = softplus
template<int ACT>
__global__ __launch_bounds__(256) void wmma_gemm_k(
    const f16* __restrict__ A, long lda, long aB,
    const f16* __restrict__ W, long ldb, long kofs, long bB,
    const float* __restrict__ bias,
    const float* __restrict__ res, long resB,
    float* __restrict__ oF, long oFB,
    f16* __restrict__ oH, long oHB,
    f16* __restrict__ oHT, long Mtot, long oHTB,
    int N, int K)
{
  __shared__ f16 As[2][BM*BKH];   // 2 x 16KB
  __shared__ f16 Bs[2][BN*BKH];   // 2 x 8KB

  const int tid  = threadIdx.x;
  const int wave = tid >> 5;
  const int lane = tid & 31;
  const int lr   = lane & 15;
  const int hs   = lane >> 4;
  const long b   = blockIdx.z;
  const int mBase = blockIdx.y * BM;
  const int nBase = blockIdx.x * BN;

  const f16* Ag = A + b*aB + (long)mBase*lda;
  const f16* Bg = W + b*bB + (long)nBase*ldb + kofs;

  // staging: 16B chunks; a 64-half row = 8 chunks; A: 1024 chunks (4/thread), B: 512 (2/thread)
  const int rowS = tid >> 3;            // 0..31
  const int colC = (tid & 7) * 8;       // 0..56 (halves)

  const unsigned aB0 = lds_off(As);
  const unsigned bB0 = lds_off(Bs);
  const unsigned aBufB = (unsigned)(BM*BKH*2);   // 16384 bytes
  const unsigned bBufB = (unsigned)(BN*BKH*2);   // 8192 bytes
  const unsigned rowOff = (unsigned)((rowS*BKH + colC)*2);

  v8f acc0 = {0.f,0.f,0.f,0.f,0.f,0.f,0.f,0.f};
  v8f acc1 = acc0, acc2 = acc0, acc3 = acc0;

  auto stage = [&](int buf, int k0){
    if (colC < K - k0){                          // EXEC-masked staging (legal: not WMMA)
      const unsigned la = aB0 + (unsigned)buf*aBufB + rowOff;
      const f16* ag = Ag + (long)rowS*lda + k0 + colC;
      async_ld_b128(la,                   ag);
      async_ld_b128(la + 32u*BKH*2,       ag + 32*lda);
      async_ld_b128(la + 64u*BKH*2,       ag + 64*lda);
      async_ld_b128(la + 96u*BKH*2,       ag + 96*lda);
      const unsigned lb = bB0 + (unsigned)buf*bBufB + rowOff;
      const f16* bg = Bg + (long)rowS*ldb + k0 + colC;
      async_ld_b128(lb,                   bg);
      async_ld_b128(lb + 32u*BKH*2,       bg + 32*ldb);
    }
  };

  stage(0, 0);                                   // prologue

  int buf = 0;
  for (int k0 = 0; k0 < K; k0 += BKH, buf ^= 1){
    wait_asynccnt0();                            // current buffer resident
    __syncthreads();                             // visible to all waves; prev compute done
    if (k0 + BKH < K) stage(buf ^ 1, k0 + BKH);  // overlap next copy with compute

    const int slices = (K - k0 >= BKH) ? 2 : 1;
    for (int s = 0; s < slices; s++){
      const f16* ap = &As[buf][(wave*16 + lr)*BKH + s*32 + hs*8];
      v8h alo = *(const v8h*)ap;
      v8h ahi = *(const v8h*)(ap + 16);
      v16h av = SHUF16(alo, ahi);
#pragma unroll
      for (int j = 0; j < 4; j++){
        const f16* bp = &Bs[buf][(j*16 + lr)*BKH + s*32 + hs*8];
        v8h blo = *(const v8h*)bp;
        v8h bhi = *(const v8h*)(bp + 16);
        v16h bv = SHUF16(blo, bhi);
        if (j == 0)      acc0 = wmma16(av, bv, acc0);
        else if (j == 1) acc1 = wmma16(av, bv, acc1);
        else if (j == 2) acc2 = wmma16(av, bv, acc2);
        else             acc3 = wmma16(av, bv, acc3);
      }
    }
  }

#pragma unroll
  for (int j = 0; j < 4; j++){
    v8f acc = (j == 0) ? acc0 : (j == 1) ? acc1 : (j == 2) ? acc2 : acc3;
    const int col = nBase + j*16 + lr;
    const float bvv = bias ? bias[col] : 0.0f;
#pragma unroll
    for (int i = 0; i < 8; i++){
      const int row = mBase + wave*16 + i + hs*8;
      const long idx = (long)row * N + col;
      float v = acc[i] + bvv;
      if (res) v += res[b*resB + idx];
      if (ACT == 1) v = qgeluf(v);
      else if (ACT == 3) v = softplusf_(v);
      if (oF)  oF[b*oFB + idx] = v;
      if (oH)  oH[b*oHB + idx] = (f16)v;
      if (oHT) oHT[b*oHTB + (long)col*Mtot + row] = (f16)v;
    }
  }
}

static void gemm_launch(hipStream_t st, int act,
    const f16* A, long lda, long aB,
    const f16* W, long ldb, long kofs, long bB,
    const float* bias, const float* res, long resB,
    float* oF, long oFB, f16* oH, long oHB,
    f16* oHT, long Mtot, long oHTB,
    int Mrows, int N, int K, int batches)
{
  dim3 g((unsigned)(N/BN), (unsigned)(Mrows/BM), (unsigned)batches);
  dim3 blk(256);
  if (act == 1)
    wmma_gemm_k<1><<<g,blk,0,st>>>(A,lda,aB,W,ldb,kofs,bB,bias,res,resB,oF,oFB,oH,oHB,oHT,Mtot,oHTB,N,K);
  else if (act == 3)
    wmma_gemm_k<3><<<g,blk,0,st>>>(A,lda,aB,W,ldb,kofs,bB,bias,res,resB,oF,oFB,oH,oHB,oHT,Mtot,oHTB,N,K);
  else
    wmma_gemm_k<0><<<g,blk,0,st>>>(A,lda,aB,W,ldb,kofs,bB,bias,res,resB,oF,oFB,oH,oHB,oHT,Mtot,oHTB,N,K);
}

// ================= windowed attention (WMMA in LDS) =================
// one block per (batch, head, window); qkv: [M x 1536] f32
__global__ __launch_bounds__(256) void attn_win_k(const float* __restrict__ qkv,
                                                  f16* __restrict__ outH, int T)
{
  const int NW = T / WINSZ;
  const int id = blockIdx.x;
  const int w  = id % NW;
  const int h  = (id / NW) % NHEADS;
  const int b  = id / (NW * NHEADS);

  __shared__ f16 Qs[WINSZ*WINSZ];
  __shared__ f16 Ks[WINSZ*WINSZ];
  __shared__ f16 Vt[WINSZ*WINSZ];
  __shared__ float Ss[WINSZ*WINSZ];

  const long rowbase = (long)b*T + (long)w*WINSZ;
  for (int idx = threadIdx.x; idx < WINSZ*WINSZ; idx += 256){
    int t = idx >> 6, d = idx & 63;
    const float* src = qkv + (rowbase + t)*(3*HIDD) + h*64 + d;
    Qs[idx]        = (f16)src[0];
    Ks[idx]        = (f16)src[HIDD];
    Vt[d*WINSZ + t] = (f16)src[2*HIDD];
  }
  __syncthreads();

  const int wave = threadIdx.x >> 5;
  const int lane = threadIdx.x & 31;
  const int lr   = lane & 15;
  const int hs   = lane >> 4;

  // scores = Q @ K^T * 1/8
  for (int rep = 0; rep < 2; rep++){
    int tile = wave*2 + rep;
    int tm = (tile >> 2) * 16, tn = (tile & 3) * 16;
    v8f acc = {0.f,0.f,0.f,0.f,0.f,0.f,0.f,0.f};
    for (int k0 = 0; k0 < WINSZ; k0 += 32){
      const f16* ap = &Qs[(tm+lr)*WINSZ + k0 + hs*8];
      v8h alo = *(const v8h*)ap;
      v8h ahi = *(const v8h*)(ap + 16);
      const f16* bp = &Ks[(tn+lr)*WINSZ + k0 + hs*8];
      v8h blo = *(const v8h*)bp;
      v8h bhi = *(const v8h*)(bp + 16);
      acc = wmma16(SHUF16(alo, ahi), SHUF16(blo, bhi), acc);
    }
#pragma unroll
    for (int i = 0; i < 8; i++)
      Ss[(tm + i + hs*8)*WINSZ + tn + lr] = acc[i] * 0.125f;
  }
  __syncthreads();

  // row softmax, write f16 probs into Qs
  if ((int)threadIdx.x < WINSZ){
    int r = threadIdx.x;
    float mx = -3.0e38f;
    for (int s = 0; s < WINSZ; s++) mx = fmaxf(mx, Ss[r*WINSZ + s]);
    float sum = 0.f;
    for (int s = 0; s < WINSZ; s++){
      float e = __expf(Ss[r*WINSZ + s] - mx);
      Ss[r*WINSZ + s] = e;
      sum += e;
    }
    float inv = 1.0f / sum;
    for (int s = 0; s < WINSZ; s++)
      Qs[r*WINSZ + s] = (f16)(Ss[r*WINSZ + s] * inv);
  }
  __syncthreads();

  // O = attn @ V  (V^T rows give B-fragments)
  for (int rep = 0; rep < 2; rep++){
    int tile = wave*2 + rep;
    int tm = (tile >> 2) * 16, tn = (tile & 3) * 16;
    v8f acc = {0.f,0.f,0.f,0.f,0.f,0.f,0.f,0.f};
    for (int k0 = 0; k0 < WINSZ; k0 += 32){
      const f16* ap = &Qs[(tm+lr)*WINSZ + k0 + hs*8];
      v8h alo = *(const v8h*)ap;
      v8h ahi = *(const v8h*)(ap + 16);
      const f16* bp = &Vt[(tn+lr)*WINSZ + k0 + hs*8];
      v8h blo = *(const v8h*)bp;
      v8h bhi = *(const v8h*)(bp + 16);
      acc = wmma16(SHUF16(alo, ahi), SHUF16(blo, bhi), acc);
    }
    const int col = h*64 + tn + lr;
#pragma unroll
    for (int i = 0; i < 8; i++){
      long row = rowbase + tm + i + hs*8;
      outH[row*HIDD + col] = (f16)acc[i];
    }
  }
}

// ================= elementwise / reduction kernels =================
__global__ void cvt_k(const float* __restrict__ a, f16* __restrict__ o, long n){
  for (long i = blockIdx.x*(long)blockDim.x + threadIdx.x; i < n; i += (long)gridDim.x*blockDim.x)
    o[i] = (f16)a[i];
}

__global__ void ew_add_k(const float* __restrict__ a, const float* __restrict__ b,
                         float* __restrict__ o, long n){
  for (long i = blockIdx.x*(long)blockDim.x + threadIdx.x; i < n; i += (long)gridDim.x*blockDim.x)
    o[i] = a[i] + b[i];
}

__global__ void add_pos_k(float* __restrict__ x, const float* __restrict__ pos, long total){
  for (long i = blockIdx.x*(long)blockDim.x + threadIdx.x; i < total; i += (long)gridDim.x*blockDim.x){
    long row = i / HIDD;
    int  c   = (int)(i % HIDD);
    int  t   = (int)(row % TLEN);
    x[i] += pos[(long)t*HIDD + c];
  }
}

__global__ void bcast_add_k(float* __restrict__ x, const float* __restrict__ v, long total){
  for (long i = blockIdx.x*(long)blockDim.x + threadIdx.x; i < total; i += (long)gridDim.x*blockDim.x){
    long b = i / ((long)TLEN*HIDD);
    int  c = (int)(i % HIDD);
    x[i] += v[b*HIDD + c];
  }
}

// per-row |feat| sum -> pad flag; motion16 = flag * (feat[t]-feat[t-1])
__global__ __launch_bounds__(256) void motion_k(const float* __restrict__ feat,
                                                f16* __restrict__ mot16){
  const long m = blockIdx.x;
  const int  t = (int)(m % TLEN);
  const float* cur = feat + m*FEATD;
  __shared__ float sm[256];
  float p = 0.f;
  for (int c = threadIdx.x; c < FEATD; c += 256) p += fabsf(cur[c]);
  float tot = block_reduce_sum(p, sm);
  const float flag = (tot != 0.0f) ? 1.0f : 0.0f;
  const float* prev = cur - FEATD;
  for (int c = threadIdx.x; c < FEATD; c += 256){
    float v = (t == 0) ? 0.0f : (cur[c] - prev[c]) * flag;
    mot16[m*FEATD + c] = (f16)v;
  }
}

__global__ __launch_bounds__(256) void layernorm_k(const float* __restrict__ x,
                                                   const float* __restrict__ g,
                                                   const float* __restrict__ bta,
                                                   float* __restrict__ oF,
                                                   f16* __restrict__ oH, int D){
  const long m = blockIdx.x;
  const float* row = x + m*D;
  __shared__ float sm[256];
  float s = 0.f, s2 = 0.f;
  for (int c = threadIdx.x; c < D; c += 256){ float v = row[c]; s += v; s2 += v*v; }
  float mean = block_reduce_sum(s, sm) / (float)D;
  float var  = block_reduce_sum(s2, sm) / (float)D - mean*mean;
  float inv  = rsqrtf(var + 1e-5f);
  for (int c = threadIdx.x; c < D; c += 256){
    float v = (row[c] - mean) * inv * g[c] + bta[c];
    if (oF) oF[m*D + c] = v;
    if (oH) oH[m*D + c] = (f16)v;
  }
}

__global__ __launch_bounds__(256) void rownorm_k(const float* __restrict__ x,
                                                 f16* __restrict__ o16, int D){
  const long m = blockIdx.x;
  const float* row = x + m*D;
  __shared__ float sm[256];
  float s = 0.f;
  for (int c = threadIdx.x; c < D; c += 256){ float v = row[c]; s += v*v; }
  float n = sqrtf(block_reduce_sum(s, sm));
  float inv = 1.0f / fmaxf(n, 1e-12f);
  for (int c = threadIdx.x; c < D; c += 256) o16[m*D + c] = (f16)(row[c]*inv);
}

// threshold(>0.5) + I, row-normalize; scores: [b][T][T] f32 -> adj f16
__global__ __launch_bounds__(256) void adj_norm_k(const float* __restrict__ s,
                                                  f16* __restrict__ adj, int T){
  const int  r = blockIdx.x % T;
  const long b = blockIdx.x / T;
  const float* row = s + (b*T + r)*(long)T;
  f16* orow = adj + (b*T + r)*(long)T;
  __shared__ float sm[256];
  float part = 0.f;
  for (int c = threadIdx.x; c < T; c += 256){
    float v = row[c];
    v = (v > 0.5f) ? v : 0.0f;
    if (c == r) v += 1.0f;
    part += v;
  }
  float tot = block_reduce_sum(part, sm);
  float inv = 1.0f / fmaxf(tot, 1e-8f);
  for (int c = threadIdx.x; c < T; c += 256){
    float v = row[c];
    v = (v > 0.5f) ? v : 0.0f;
    if (c == r) v += 1.0f;
    orow[c] = (f16)(v * inv);
  }
}

__global__ void dadj_k(f16* __restrict__ adj){
  for (long i = blockIdx.x*(long)blockDim.x + threadIdx.x; i < (long)TLEN*TLEN;
       i += (long)gridDim.x*blockDim.x){
    int t = (int)(i / TLEN), s = (int)(i % TLEN);
    adj[i] = (f16)__expf(-fabsf((float)(t - s)) * 0.36787944117144233f); // exp(-|i-j|/e)
  }
}

// video-pos MLP: per batch, l1 (1->128, qgelu) then l2 (128->512)
__global__ __launch_bounds__(128) void vpp_k(const float* __restrict__ vp,
                                             const float* __restrict__ w1, const float* __restrict__ b1,
                                             const float* __restrict__ w2, const float* __restrict__ b2,
                                             float* __restrict__ vout){
  const int b = blockIdx.x;
  __shared__ float h[128];
  h[threadIdx.x] = qgeluf(w1[threadIdx.x]*vp[b] + b1[threadIdx.x]);
  __syncthreads();
  for (int o = threadIdx.x; o < HIDD; o += 128){
    float s = b2[o];
    const float* wr = w2 + (long)o*128;
    for (int j = 0; j < 128; j++) s += wr[j]*h[j];
    vout[(long)b*HIDD + o] = s;
  }
}

// mamba causal depthwise conv (dir=0 fwd, dir=1 time-reversed) + SiLU
__global__ void mamba_conv_k(const float* __restrict__ minp, // [M x 1024], xm in cols 0..511
                             const float* __restrict__ convw, const float* __restrict__ convb,
                             float* __restrict__ xc, f16* __restrict__ xc16, int dir){
  long i = blockIdx.x*(long)blockDim.x + threadIdx.x;
  if (i >= (long)MROWS*HIDD) return;
  int  d  = (int)(i % HIDD);
  long bt = i / HIDD;
  int  t  = (int)(bt % TLEN);
  long b  = bt / TLEN;
  float s = convb[d];
#pragma unroll
  for (int k = 0; k < 4; k++){
    int tt = (dir == 0) ? (t - 3 + k) : (t + 3 - k);
    if (tt >= 0 && tt < TLEN) s += convw[d*4 + k] * minp[(b*TLEN + tt)*(2*HIDD) + d];
  }
  s = siluf(s);
  xc[i] = s;
  xc16[i] = (f16)s;
}

// selective scan: one thread per (b, d)
__global__ void mamba_scan_k(const float* __restrict__ delta, const float* __restrict__ xc,
                             const float* __restrict__ dbc,   // [M x 64]: dt|B|C
                             const float* __restrict__ minp,  // z at cols 512..1023
                             const float* __restrict__ A_log, const float* __restrict__ Dp,
                             f16* __restrict__ g16, int dir){
  int i = blockIdx.x*blockDim.x + threadIdx.x;
  if (i >= BATCH*HIDD) return;
  int d = i % HIDD;
  int b = i / HIDD;
  float Ar[NSTATE], h[NSTATE];
#pragma unroll
  for (int n = 0; n < NSTATE; n++){ Ar[n] = -__expf(A_log[d*NSTATE + n]); h[n] = 0.f; }
  const float Dv = Dp[d];
  for (int s = 0; s < TLEN; s++){
    int t = (dir == 0) ? s : (TLEN - 1 - s);
    long bt = (long)b*TLEN + t;
    float dl = delta[bt*HIDD + d];
    float xv = xc[bt*HIDD + d];
    const float* bc = dbc + bt*64;
    float y = 0.f;
#pragma unroll
    for (int n = 0; n < NSTATE; n++){
      float dA = __expf(dl * Ar[n]);
      h[n] = dA*h[n] + dl * bc[DTRANK + n] * xv;
      y += h[n] * bc[DTRANK + NSTATE + n];
    }
    float z = minp[bt*(2*HIDD) + HIDD + d];
    g16[bt*HIDD + d] = (f16)((y + xv*Dv) * siluf(z));
  }
}

// small dense layer (heads l2): one thread per output
__global__ void small_linear_k(const float* __restrict__ A, const float* __restrict__ W,
                               const float* __restrict__ bias, float* __restrict__ o,
                               long Mr, int N, int K){
  long i = blockIdx.x*(long)blockDim.x + threadIdx.x;
  if (i >= Mr*(long)N) return;
  long m = i / N;
  int  n = (int)(i % N);
  float s = bias ? bias[n] : 0.f;
  const float* a = A + m*K;
  const float* w = W + (long)n*K;
  for (int k = 0; k < K; k++) s += a[k]*w[k];
  o[i] = s;
}

__global__ __launch_bounds__(256) void proto_norm_k(const float* __restrict__ pr,
                                                    float* __restrict__ prn){
  const int j = blockIdx.x;
  __shared__ float sm[256];
  float s = 0.f;
  for (int c = threadIdx.x; c < FEATD; c += 256){ float v = pr[(long)j*FEATD + c]; s += v*v; }
  float n = sqrtf(block_reduce_sum(s, sm));
  float inv = 1.0f / fmaxf(n, 1e-12f);
  for (int c = threadIdx.x; c < FEATD; c += 256) prn[(long)j*FEATD + c] = pr[(long)j*FEATD + c]*inv;
}

// path *= 1 + sigmoid(pw) * sigmoid(<rf_n, pr_n_j>)
__global__ __launch_bounds__(256) void path_gate_k(const float* __restrict__ raw,
                                                   const float* __restrict__ prn,
                                                   const float* __restrict__ pw,
                                                   float* __restrict__ path){
  const long m = blockIdx.x;
  const float* r = raw + m*FEATD;
  __shared__ float sm[256];
  float p = 0.f;
  for (int c = threadIdx.x; c < FEATD; c += 256) p += r[c]*r[c];
  float nn = sqrtf(block_reduce_sum(p, sm));
  float invn = 1.0f / fmaxf(nn, 1e-12f);
  float sw = sigmf(pw[0]);
  for (int j = 0; j < NPATH; j++){
    float part = 0.f;
    for (int c = threadIdx.x; c < FEATD; c += 256) part += r[c]*prn[(long)j*FEATD + c];
    float dot = block_reduce_sum(part, sm);
    if (threadIdx.x == 0){
      float gate = sigmf(dot * invn);
      path[m*NPATH + j] *= (1.0f + sw*gate);
    }
  }
}

// ================= host orchestration =================
extern "C" void kernel_launch(void* const* d_in, const int* in_sizes, int n_in,
                              void* d_out, int out_size, void* d_ws, size_t ws_size,
                              hipStream_t stream)
{
  (void)in_sizes; (void)n_in; (void)out_size; (void)ws_size;
  const float* features = (const float*)d_in[0];
  const float* rawf     = (const float*)d_in[1];
  const float* vposr    = (const float*)d_in[2];
  auto P = [&](int i)->const float*{ return (const float*)d_in[i]; };

  // parameter indices (setup_inputs() dict order)
  const int IP_W=3, IP_B=4, POS=5, MP_W=6, MP_B=7;
  const int BLK0=8;                                  // 2 blocks x 12 entries
  const int GS1_W=32, GS1_B=33, GS1_RW=34, GS1_RB=35, GS2_W=36, GS2_B=37;
  const int GD1_W=38, GD1_B=39, GD1_RW=40, GD1_RB=41, GD2_W=42, GD2_B=43;
  const int AGP_W=44, AGP_B=45, RFP_W=46, RFP_B=47, GCP_W=48, GCP_B=49;
  const int VP_L1W=50, VP_L1B=51, VP_L2W=52, VP_L2B=53;
  const int MF=54, MB_=63;                           // in_proj,conv_w,conv_b,x_proj,dt_w,dt_b,A_log,D,out_proj
  const int BP_W=72, BP_B=73;
  const int AH=74, PH=80;                            // lng,lnb,l1w,l1b,l2w,l2b
  const int PROTO=86, PW=87;

  size_t off = 0;
  auto alloc = [&](size_t bytes)->void*{
    void* p = (char*)d_ws + off;
    off += (bytes + 255) & ~(size_t)255;
    return p;
  };
  auto tohalf = [&](const float* src, long n)->f16*{
    f16* d = (f16*)alloc((size_t)n*sizeof(f16));
    cvt_k<<<dim3((unsigned)((n + 1023)/1024)), dim3(256), 0, stream>>>(src, d, n);
    return d;
  };

  // ---- weights -> f16 ----
  f16* ipw16  = tohalf(P(IP_W), 512L*1024);
  f16* mpw16  = tohalf(P(MP_W), 512L*1024);
  f16 *qkvw16[2], *outw16[2], *ff1w16[2], *ff2w16[2];
  for (int l = 0; l < 2; l++){
    int base = BLK0 + l*12;
    qkvw16[l] = tohalf(P(base+4),  1536L*512);
    outw16[l] = tohalf(P(base+6),  512L*512);
    ff1w16[l] = tohalf(P(base+8),  2048L*512);
    ff2w16[l] = tohalf(P(base+10), 512L*2048);
  }
  f16* gs1w16  = tohalf(P(GS1_W),  256L*512);
  f16* gs1rw16 = tohalf(P(GS1_RW), 256L*512);
  f16* gs2w16  = tohalf(P(GS2_W),  256L*256);
  f16* gd1w16  = tohalf(P(GD1_W),  256L*512);
  f16* gd1rw16 = tohalf(P(GD1_RW), 256L*512);
  f16* gd2w16  = tohalf(P(GD2_W),  256L*256);
  f16* agpw16  = tohalf(P(AGP_W),  512L*512);
  f16* rfpw16  = tohalf(P(RFP_W),  512L*1024);
  f16* gcpw16  = tohalf(P(GCP_W),  512L*512);
  f16 *inpw16[2], *xprw16[2], *dtw16[2], *outpw16[2];
  for (int d = 0; d < 2; d++){
    int mb = (d == 0) ? MF : MB_;
    inpw16[d]  = tohalf(P(mb+0), 1024L*512);
    xprw16[d]  = tohalf(P(mb+3), 64L*512);
    dtw16[d]   = tohalf(P(mb+4), 512L*32);
    outpw16[d] = tohalf(P(mb+8), 512L*512);
  }
  f16* bpw16   = tohalf(P(BP_W), 512L*1536);
  f16* ahl1w16 = tohalf(P(AH+2), 256L*512);
  f16* phl1w16 = tohalf(P(PH+2), 256L*512);

  // ---- activation buffers ----
  f16*  feat16 = tohalf(features, (long)MROWS*FEATD);
  f16*  raw16  = tohalf(rawf,     (long)MROWS*FEATD);
  f16*  mot16  = (f16*)alloc((size_t)MROWS*FEATD*2);
  float* x     = (float*)alloc((size_t)MROWS*HIDD*4);
  f16*  x16    = (f16*)alloc((size_t)MROWS*HIDD*2);
  float* tA    = (float*)alloc((size_t)MROWS*HIDD*4);
  float* tB    = (float*)alloc((size_t)MROWS*HIDD*4);
  f16*  h16    = (f16*)alloc((size_t)MROWS*HIDD*2);
  float* big   = (float*)alloc((size_t)MROWS*3*HIDD*4);  // reused: qkv / sim scores / mamba in_proj
  f16*  attn16 = (f16*)alloc((size_t)MROWS*HIDD*2);
  f16*  fft16  = (f16*)alloc((size_t)MROWS*4*HIDD*2);
  float* vout  = (float*)alloc((size_t)BATCH*HIDD*4);
  f16*  xn16   = (f16*)alloc((size_t)MROWS*HIDD*2);
  f16*  adj16  = (f16*)alloc((size_t)BATCH*TLEN*TLEN*2); // sim adj, then pathology adj
  f16*  dadj16 = (f16*)alloc((size_t)TLEN*TLEN*2);
  f16*  supT16 = (f16*)alloc((size_t)MROWS*HIDD*2);      // [N x 4096] transposed supports
  float* R     = (float*)alloc((size_t)MROWS*GCND*4);
  float* xs    = (float*)alloc((size_t)MROWS*GCND*4);
  f16*  xs16   = (f16*)alloc((size_t)MROWS*GCND*2);
  float* xd    = (float*)alloc((size_t)MROWS*GCND*4);
  f16*  xd16   = (f16*)alloc((size_t)MROWS*GCND*2);
  float* x_a   = (float*)alloc((size_t)MROWS*HIDD*4);
  f16*  x_a16  = (f16*)alloc((size_t)MROWS*HIDD*2);
  float* xc    = (float*)alloc((size_t)MROWS*HIDD*4);
  f16*  xc16   = (f16*)alloc((size_t)MROWS*HIDD*2);
  float* dbc   = (float*)alloc((size_t)MROWS*64*4);
  f16*  dbc16  = (f16*)alloc((size_t)MROWS*64*2);
  float* delta = (float*)alloc((size_t)MROWS*HIDD*4);
  f16*  g16    = (f16*)alloc((size_t)MROWS*HIDD*2);
  f16*  hf16[2];
  hf16[0] = (f16*)alloc((size_t)MROWS*HIDD*2);
  hf16[1] = (f16*)alloc((size_t)MROWS*HIDD*2);
  float* xctx  = (float*)alloc((size_t)MROWS*HIDD*4);
  float* l1out = (float*)alloc((size_t)MROWS*GCND*4);
  float* x_pb  = (float*)alloc((size_t)MROWS*HIDD*4);
  f16*  x_pb16 = (f16*)alloc((size_t)MROWS*HIDD*2);
  float* prn   = (float*)alloc((size_t)NPATH*FEATD*4);

  const long MH = (long)MROWS*HIDD;

  // ---- input embedding: x = ip(features) + pos + mp(motion*notpad) ----
  motion_k<<<MROWS, 256, 0, stream>>>(features, mot16);
  gemm_launch(stream, 0, feat16, FEATD,0, ipw16, FEATD,0,0, P(IP_B), nullptr,0,
              tA,0, nullptr,0, nullptr,0,0, MROWS, HIDD, FEATD, 1);
  add_pos_k<<<2048, 256, 0, stream>>>(tA, P(POS), MH);
  gemm_launch(stream, 0, mot16, FEATD,0, mpw16, FEATD,0,0, P(MP_B), tA,0,
              x,0, nullptr,0, nullptr,0,0, MROWS, HIDD, FEATD, 1);

  // ---- transformer blocks (windowed attention) ----
  for (int l = 0; l < 2; l++){
    int base = BLK0 + l*12;
    layernorm_k<<<MROWS, 256, 0, stream>>>(x, P(base+0), P(base+1), nullptr, h16, HIDD);
    gemm_launch(stream, 0, h16, HIDD,0, qkvw16[l], HIDD,0,0, P(base+5), nullptr,0,
                big,0, nullptr,0, nullptr,0,0, MROWS, 3*HIDD, HIDD, 1);
    attn_win_k<<<BATCH*NHEADS*(TLEN/WINSZ), 256, 0, stream>>>(big, attn16, TLEN);
    gemm_launch(stream, 0, attn16, HIDD,0, outw16[l], HIDD,0,0, P(base+7), x,0,
                x,0, nullptr,0, nullptr,0,0, MROWS, HIDD, HIDD, 1);
    layernorm_k<<<MROWS, 256, 0, stream>>>(x, P(base+2), P(base+3), nullptr, h16, HIDD);
    gemm_launch(stream, 1, h16, HIDD,0, ff1w16[l], HIDD,0,0, P(base+9), nullptr,0,
                nullptr,0, fft16,0, nullptr,0,0, MROWS, 4*HIDD, HIDD, 1);
    gemm_launch(stream, 0, fft16, 4*HIDD,0, ff2w16[l], 4*HIDD,0,0, P(base+11), x,0,
                x,0, nullptr,0, nullptr,0,0, MROWS, HIDD, 4*HIDD, 1);
  }

  // ---- video position MLP add ----
  vpp_k<<<BATCH, 128, 0, stream>>>(vposr, P(VP_L1W), P(VP_L1B), P(VP_L2W), P(VP_L2B), vout);
  bcast_add_k<<<2048, 256, 0, stream>>>(x, vout, MH);
  cvt_k<<<4096, 256, 0, stream>>>(x, x16, MH);

  // ---- similarity adjacency ----
  rownorm_k<<<MROWS, 256, 0, stream>>>(x, xn16, HIDD);
  gemm_launch(stream, 0, xn16, HIDD,(long)TLEN*HIDD, xn16, HIDD,0,(long)TLEN*HIDD,
              nullptr, nullptr,0, big,(long)TLEN*TLEN, nullptr,0, nullptr,0,0,
              TLEN, TLEN, HIDD, BATCH);
  adj_norm_k<<<BATCH*TLEN, 256, 0, stream>>>(big, adj16, TLEN);
  dadj_k<<<2048, 256, 0, stream>>>(dadj16);

  // ---- GCN (similarity branch) ----
  gemm_launch(stream, 0, x16, HIDD,0, gs1rw16, HIDD,0,0, P(GS1_RB), nullptr,0,
              R,0, nullptr,0, nullptr,0,0, MROWS, GCND, HIDD, 1);
  gemm_launch(stream, 0, x16, HIDD,0, gs1w16, HIDD,0,0, nullptr, nullptr,0,
              nullptr,0, nullptr,0, supT16, MROWS,0, MROWS, GCND, HIDD, 1);
  gemm_launch(stream, 1, adj16, TLEN,(long)TLEN*TLEN, supT16, MROWS,0,TLEN, P(GS1_B),
              R,(long)TLEN*GCND, xs,(long)TLEN*GCND, xs16,(long)TLEN*GCND, nullptr,0,0,
              TLEN, GCND, TLEN, BATCH);
  gemm_launch(stream, 0, xs16, GCND,0, gs2w16, GCND,0,0, nullptr, nullptr,0,
              nullptr,0, nullptr,0, supT16, MROWS,0, MROWS, GCND, GCND, 1);
  gemm_launch(stream, 1, adj16, TLEN,(long)TLEN*TLEN, supT16, MROWS,0,TLEN, P(GS2_B),
              xs,(long)TLEN*GCND, xs,(long)TLEN*GCND, xs16,(long)TLEN*GCND, nullptr,0,0,
              TLEN, GCND, TLEN, BATCH);

  // ---- GCN (distance branch; shared dadj => A batch stride 0) ----
  gemm_launch(stream, 0, x16, HIDD,0, gd1rw16, HIDD,0,0, P(GD1_RB), nullptr,0,
              R,0, nullptr,0, nullptr,0,0, MROWS, GCND, HIDD, 1);
  gemm_launch(stream, 0, x16, HIDD,0, gd1w16, HIDD,0,0, nullptr, nullptr,0,
              nullptr,0, nullptr,0, supT16, MROWS,0, MROWS, GCND, HIDD, 1);
  gemm_launch(stream, 1, dadj16, TLEN,0, supT16, MROWS,0,TLEN, P(GD1_B),
              R,(long)TLEN*GCND, xd,(long)TLEN*GCND, xd16,(long)TLEN*GCND, nullptr,0,0,
              TLEN, GCND, TLEN, BATCH);
  gemm_launch(stream, 0, xd16, GCND,0, gd2w16, GCND,0,0, nullptr, nullptr,0,
              nullptr,0, nullptr,0, supT16, MROWS,0, MROWS, GCND, GCND, 1);
  gemm_launch(stream, 1, dadj16, TLEN,0, supT16, MROWS,0,TLEN, P(GD2_B),
              xd,(long)TLEN*GCND, xd,(long)TLEN*GCND, xd16,(long)TLEN*GCND, nullptr,0,0,
              TLEN, GCND, TLEN, BATCH);

  // ---- x_a = anatomy_gcn_proj(concat(xs, xd)) ----
  gemm_launch(stream, 0, xs16, GCND,0, agpw16, HIDD,0,0, nullptr, nullptr,0,
              tA,0, nullptr,0, nullptr,0,0, MROWS, HIDD, GCND, 1);
  gemm_launch(stream, 0, xd16, GCND,0, agpw16, HIDD,GCND,0, P(AGP_B), tA,0,
              x_a,0, x_a16,0, nullptr,0,0, MROWS, HIDD, GCND, 1);

  // ---- bidirectional mamba ----
  for (int dir = 0; dir < 2; dir++){
    int mb = (dir == 0) ? MF : MB_;
    gemm_launch(stream, 0, x_a16, HIDD,0, inpw16[dir], HIDD,0,0, nullptr, nullptr,0,
                big,0, nullptr,0, nullptr,0,0, MROWS, 2*HIDD, HIDD, 1);
    mamba_conv_k<<<(MROWS*HIDD + 255)/256, 256, 0, stream>>>(big, P(mb+1), P(mb+2), xc, xc16, dir);
    gemm_launch(stream, 0, xc16, HIDD,0, xprw16[dir], HIDD,0,0, nullptr, nullptr,0,
                dbc,0, dbc16,0, nullptr,0,0, MROWS, 64, HIDD, 1);
    gemm_launch(stream, 3, dbc16, 64,0, dtw16[dir], DTRANK,0,0, P(mb+5), nullptr,0,
                delta,0, nullptr,0, nullptr,0,0, MROWS, HIDD, DTRANK, 1);
    mamba_scan_k<<<(BATCH*HIDD + 255)/256, 256, 0, stream>>>(delta, xc, dbc, big,
                                                             P(mb+6), P(mb+7), g16, dir);
    gemm_launch(stream, 0, g16, HIDD,0, outpw16[dir], HIDD,0,0, nullptr, nullptr,0,
                nullptr,0, hf16[dir],0, nullptr,0,0, MROWS, HIDD, HIDD, 1);
  }

  // ---- boundary projection: concat(x_a, h_fwd, h_bwd) -> x_ctx ----
  gemm_launch(stream, 0, x_a16, HIDD,0, bpw16, 3*HIDD,0,0, nullptr, nullptr,0,
              tA,0, nullptr,0, nullptr,0,0, MROWS, HIDD, HIDD, 1);
  gemm_launch(stream, 0, hf16[0], HIDD,0, bpw16, 3*HIDD,HIDD,0, nullptr, tA,0,
              tB,0, nullptr,0, nullptr,0,0, MROWS, HIDD, HIDD, 1);
  gemm_launch(stream, 0, hf16[1], HIDD,0, bpw16, 3*HIDD,2*HIDD,0, P(BP_B), tB,0,
              xctx,0, nullptr,0, nullptr,0,0, MROWS, HIDD, HIDD, 1);

  // ---- anatomy head ----
  layernorm_k<<<MROWS, 256, 0, stream>>>(xctx, P(AH+0), P(AH+1), nullptr, h16, HIDD);
  gemm_launch(stream, 1, h16, HIDD,0, ahl1w16, HIDD,0,0, P(AH+3), nullptr,0,
              l1out,0, nullptr,0, nullptr,0,0, MROWS, GCND, HIDD, 1);
  small_linear_k<<<((long)MROWS*NANAT + 255)/256, 256, 0, stream>>>(
      l1out, P(AH+4), P(AH+5), (float*)d_out, MROWS, NANAT, GCND);

  // ---- pathology branch ----
  gemm_launch(stream, 0, raw16, FEATD,0, rfpw16, FEATD,0,0, P(RFP_B), x_a,0,
              x_pb,0, x_pb16,0, nullptr,0,0, MROWS, HIDD, FEATD, 1);
  rownorm_k<<<MROWS, 256, 0, stream>>>(x_pb, xn16, HIDD);
  gemm_launch(stream, 0, xn16, HIDD,(long)TLEN*HIDD, xn16, HIDD,0,(long)TLEN*HIDD,
              nullptr, nullptr,0, big,(long)TLEN*TLEN, nullptr,0, nullptr,0,0,
              TLEN, TLEN, HIDD, BATCH);
  adj_norm_k<<<BATCH*TLEN, 256, 0, stream>>>(big, adj16, TLEN);
  gemm_launch(stream, 0, x_pb16, HIDD,0, gcpw16, HIDD,0,0, nullptr, nullptr,0,
              nullptr,0, nullptr,0, supT16, MROWS,0, MROWS, HIDD, HIDD, 1);
  gemm_launch(stream, 1, adj16, TLEN,(long)TLEN*TLEN, supT16, MROWS,0,TLEN, P(GCP_B),
              x_pb,(long)TLEN*HIDD, tA,(long)TLEN*HIDD, nullptr,0, nullptr,0,0,
              TLEN, HIDD, TLEN, BATCH);
  ew_add_k<<<2048, 256, 0, stream>>>(tA, x_pb, tB, MH);   // x_p
  layernorm_k<<<MROWS, 256, 0, stream>>>(tB, P(PH+0), P(PH+1), nullptr, h16, HIDD);
  gemm_launch(stream, 1, h16, HIDD,0, phl1w16, HIDD,0,0, P(PH+3), nullptr,0,
              l1out,0, nullptr,0, nullptr,0,0, MROWS, GCND, HIDD, 1);
  float* path = (float*)d_out + (long)MROWS*NANAT;
  small_linear_k<<<((long)MROWS*NPATH + 255)/256, 256, 0, stream>>>(
      l1out, P(PH+4), P(PH+5), path, MROWS, NPATH, GCND);

  // ---- prototype gating ----
  proto_norm_k<<<NPATH, 256, 0, stream>>>(P(PROTO), prn);
  path_gate_k<<<MROWS, 256, 0, stream>>>(rawf, prn, P(PW), path);
}
